// GATLayer_17789754540237
// MI455X (gfx1250) — matrix operand order repaired and checked by reading, not compiled, not run on hardware
//
#include <hip/hip_runtime.h>
#include <hip/hip_bf16.h>

#define NUM_HEADS 8
#define HEAD_DIM 16
#define D 128
#define NEG_SLOPE 0.2f

typedef __attribute__((ext_vector_type(16))) __bf16 v16bf;
typedef __attribute__((ext_vector_type(8)))  float  v8f;
typedef __attribute__((ext_vector_type(4)))  float  v4f;

// ---- order-preserving uint encoding of float (for atomic segment-max) ----
__device__ __forceinline__ unsigned ordkey(float f) {
    unsigned u = __float_as_uint(f);
    return (u & 0x80000000u) ? ~u : (u ^ 0x80000000u);
}
__device__ __forceinline__ float ordval(unsigned k) {
    unsigned u = (k & 0x80000000u) ? (k ^ 0x80000000u) : ~k;
    return __uint_as_float(u);
}

// ---- K0: split W (fp32 128x128) into bf16 hi/lo ----
__global__ void k_convert_W(const float* __restrict__ W,
                            __bf16* __restrict__ Whi, __bf16* __restrict__ Wlo) {
    int i = blockIdx.x * blockDim.x + threadIdx.x;  // 16384 threads
    float w = W[i];
    __bf16 hi = (__bf16)w;
    Whi[i] = hi;
    Wlo[i] = (__bf16)(w - (float)hi);
}

// ---- K1: h = x @ W^T via v_wmma_f32_16x16x32_bf16, bf16 hi/lo split ----
// One wave per 16-row slab; wave covers all 8 N-tiles (16x128 output).
__global__ void __launch_bounds__(256) k_gemm(const float* __restrict__ x,
                                              const __bf16* __restrict__ Whi,
                                              const __bf16* __restrict__ Wlo,
                                              float* __restrict__ h,
                                              int nSlabs) {
    int wave = (int)((blockIdx.x * blockDim.x + threadIdx.x) >> 5);
    int lane = threadIdx.x & 31;
    if (wave >= nSlabs) return;               // wave-uniform: EXEC all-ones inside

    int m0      = wave * 16;
    int halfsel = lane >> 4;                  // 0: lanes 0-15, 1: lanes 16-31
    int l16     = lane & 15;
    const float* xrow = x + (size_t)(m0 + l16) * D;

    v8f acc[8] = {};

    #pragma unroll
    for (int ks = 0; ks < 4; ks++) {          // K = 128 in 4 steps of 32
        int kb = ks * 32;
        // A fragment: contiguous K order (element i = K 16*halfsel + i), same as B.
        // lanes 0-15 hold K in {0..15}; lanes 16-31 hold {16..31}
        int ka = kb + halfsel * 16;
        v16bf ahi, alo;
        #pragma unroll
        for (int i = 0; i < 8; i++) {
            float f0 = xrow[ka + i];
            float f1 = xrow[ka + 8 + i];
            __bf16 h0 = (__bf16)f0, h1 = (__bf16)f1;
            ahi[i]     = h0;
            ahi[8 + i] = h1;
            alo[i]     = (__bf16)(f0 - (float)h0);
            alo[8 + i] = (__bf16)(f1 - (float)h1);
        }
        // B fragment: lane holds 16 consecutive K of column o = t*16 + l16
        // (column o of B == row o of W, contiguous bf16 in memory)
        int kbB = kb + halfsel * 16;
        #pragma unroll
        for (int t = 0; t < 8; t++) {
            int o = t * 16 + l16;
            v16bf bhi = *(const v16bf*)(Whi + (size_t)o * D + kbB);
            v16bf blo = *(const v16bf*)(Wlo + (size_t)o * D + kbB);
            acc[t] = __builtin_amdgcn_wmma_f32_16x16x32_bf16(
                         false, ahi, false, bhi, (short)0, acc[t], false, false);
            acc[t] = __builtin_amdgcn_wmma_f32_16x16x32_bf16(
                         false, alo, false, bhi, (short)0, acc[t], false, false);
            acc[t] = __builtin_amdgcn_wmma_f32_16x16x32_bf16(
                         false, ahi, false, blo, (short)0, acc[t], false, false);
        }
    }
    // C/D layout: VGPR r -> M = r + 8*halfsel, N = l16
    int mbase = m0 + halfsel * 8;
    #pragma unroll
    for (int t = 0; t < 8; t++) {
        int o = t * 16 + l16;
        #pragma unroll
        for (int r = 0; r < 8; r++)
            h[(size_t)(mbase + r) * D + o] = acc[t][r];
    }
}

// ---- K2: per-(node,head) attention logits s,d; init m-key, z, zero out ----
__global__ void k_attn(const float* __restrict__ hfeat,
                       const float* __restrict__ sa, const float* __restrict__ da,
                       float* __restrict__ s, float* __restrict__ dl,
                       unsigned* __restrict__ mkey, float* __restrict__ z,
                       float* __restrict__ out, int N) {
    int tid = blockIdx.x * blockDim.x + threadIdx.x;
    if (tid >= N * NUM_HEADS) return;
    int n  = tid / NUM_HEADS;
    int hd = tid % NUM_HEADS;
    const float* hp = hfeat + (size_t)n * D + hd * HEAD_DIM;
    float ssum = 0.f, dsum = 0.f;
    #pragma unroll
    for (int k = 0; k < HEAD_DIM; k++) {
        float v = hp[k];
        ssum += v * sa[hd * HEAD_DIM + k];
        dsum += v * da[hd * HEAD_DIM + k];
    }
    s[tid]  = ssum;
    dl[tid] = dsum;
    mkey[tid] = ordkey(-__builtin_inff());
    z[tid]    = 0.f;
    float* op = out + (size_t)n * D + hd * HEAD_DIM;
    #pragma unroll
    for (int k = 0; k < HEAD_DIM; k++) op[k] = 0.f;
}

__device__ __forceinline__ void edge_rc(const int* ei, int e, int E, int& row, int& col) {
    if (e < E) { row = ei[e]; col = ei[E + e]; }
    else       { row = col = e - E; }          // self loops appended
}

// ---- K3: segment max of leaky-relu logits via atomic u32 max ----
__global__ void k_edge_max(const int* __restrict__ ei,
                           const float* __restrict__ s, const float* __restrict__ dl,
                           unsigned* __restrict__ mkey, int E, int N) {
    long tid = (long)blockIdx.x * blockDim.x + threadIdx.x;
    long tot = (long)(E + N) * NUM_HEADS;
    if (tid >= tot) return;
    int e  = (int)(tid >> 3);
    int hd = (int)(tid & 7);
    int row, col; edge_rc(ei, e, E, row, col);
    float v = s[row * NUM_HEADS + hd] + dl[col * NUM_HEADS + hd];
    v = (v >= 0.f) ? v : NEG_SLOPE * v;
    atomicMax(mkey + row * NUM_HEADS + hd, ordkey(v));
}

// ---- K4: softmax denominators z via atomic fp32 add ----
__global__ void k_edge_sum(const int* __restrict__ ei,
                           const float* __restrict__ s, const float* __restrict__ dl,
                           const unsigned* __restrict__ mkey, float* __restrict__ z,
                           int E, int N) {
    long tid = (long)blockIdx.x * blockDim.x + threadIdx.x;
    long tot = (long)(E + N) * NUM_HEADS;
    if (tid >= tot) return;
    int e  = (int)(tid >> 3);
    int hd = (int)(tid & 7);
    int row, col; edge_rc(ei, e, E, row, col);
    int idx = row * NUM_HEADS + hd;
    float v = s[idx] + dl[col * NUM_HEADS + hd];
    v = (v >= 0.f) ? v : NEG_SLOPE * v;
    float w = __expf(v - ordval(mkey[idx]));
    atomicAdd(z + idx, w);
}

// ---- K5: SpMM scatter: out[row] += a * h[col]; one wave32 per edge ----
__global__ void __launch_bounds__(256) k_spmm(const int* __restrict__ ei,
                                              const float* __restrict__ s,
                                              const float* __restrict__ dl,
                                              const unsigned* __restrict__ mkey,
                                              const float* __restrict__ z,
                                              const float* __restrict__ hfeat,
                                              float* __restrict__ out, int E, int N) {
    long wave = ((long)blockIdx.x * blockDim.x + threadIdx.x) >> 5;
    int  lane = threadIdx.x & 31;
    if (wave >= (long)(E + N)) return;
    int e = (int)wave;
    int row, col; edge_rc(ei, e, E, row, col);
    int hd  = lane >> 2;                       // 4 lanes per head (4 feats each)
    int idx = row * NUM_HEADS + hd;
    float v = s[idx] + dl[col * NUM_HEADS + hd];
    v = (v >= 0.f) ? v : NEG_SLOPE * v;
    float a = __expf(v - ordval(mkey[idx])) / z[idx];
    v4f hv = *(const v4f*)(hfeat + (size_t)col * D + lane * 4);
    float* op = out + (size_t)row * D + lane * 4;
    atomicAdd(op + 0, a * hv.x);
    atomicAdd(op + 1, a * hv.y);
    atomicAdd(op + 2, a * hv.z);
    atomicAdd(op + 3, a * hv.w);
}

extern "C" void kernel_launch(void* const* d_in, const int* in_sizes, int n_in,
                              void* d_out, int out_size, void* d_ws, size_t ws_size,
                              hipStream_t stream) {
    const float* x  = (const float*)d_in[0];
    const int*   ei = (const int*)  d_in[1];
    const float* W  = (const float*)d_in[2];
    const float* sa = (const float*)d_in[3];
    const float* da = (const float*)d_in[4];
    const int N = in_sizes[0] / D;     // 100000
    const int E = in_sizes[1] / 2;     // 1600000
    float* out = (float*)d_out;

    // workspace layout
    char* p = (char*)d_ws;
    __bf16* Whi = (__bf16*)p;  p += (size_t)D * D * 2;
    __bf16* Wlo = (__bf16*)p;  p += (size_t)D * D * 2;
    float* h    = (float*)p;   p += (size_t)N * D * 4;
    float* s    = (float*)p;   p += (size_t)N * NUM_HEADS * 4;
    float* dl   = (float*)p;   p += (size_t)N * NUM_HEADS * 4;
    unsigned* mkey = (unsigned*)p; p += (size_t)N * NUM_HEADS * 4;
    float* z    = (float*)p;

    k_convert_W<<<(D * D) / 256, 256, 0, stream>>>(W, Whi, Wlo);

    int nSlabs = N / 16;                                   // N divisible by 16
    k_gemm<<<(nSlabs + 7) / 8, 256, 0, stream>>>(x, Whi, Wlo, h, nSlabs);

    long nh = (long)N * NUM_HEADS;
    k_attn<<<(unsigned)((nh + 255) / 256), 256, 0, stream>>>(h, sa, da, s, dl, mkey, z, out, N);

    long em = (long)(E + N) * NUM_HEADS;
    unsigned eb = (unsigned)((em + 255) / 256);
    k_edge_max<<<eb, 256, 0, stream>>>(ei, s, dl, mkey, E, N);
    k_edge_sum<<<eb, 256, 0, stream>>>(ei, s, dl, mkey, z, E, N);

    long sw = (long)(E + N) * 32;
    k_spmm<<<(unsigned)((sw + 255) / 256), 256, 0, stream>>>(ei, s, dl, mkey, z, h, out, E, N);
}
